// BertSelfAttention_41549513622119
// MI455X (gfx1250) — matrix operand order repaired and checked by reading, not compiled, not run on hardware
//
#include <hip/hip_runtime.h>
#include <hip/hip_bf16.h>

#define HIDC 128
#define NHEAD 8
#define HD 16

typedef __attribute__((ext_vector_type(16))) __bf16    v16bf;
typedef __attribute__((ext_vector_type(8)))  float     v8f;
typedef __attribute__((ext_vector_type(8)))  unsigned  v8u;

// pack two fp32 into one dword of two bf16 (lowers to v_cvt_pk_bf16_f32)
__device__ __forceinline__ unsigned pk_bf16(float lo, float hi)
{
    const unsigned short l = __builtin_bit_cast(unsigned short, (__bf16)lo);
    const unsigned short h = __builtin_bit_cast(unsigned short, (__bf16)hi);
    return (unsigned)l | ((unsigned)h << 16);
}

// ---------------------------------------------------------------------------
// Kernel 0: pre-swizzle Wq|Wk|Wv (fp32 row-major 128x128) into the WMMA
// B-fragment register layout, bf16-packed.  Fragment f = (m*4 + k32)*8 + ct
// occupies 32 lanes x 8 dwords (1 KB); lane (half=l>>4, mrow=l&15), dword v
// packs B[k][n] pair with k = k32*32 + half*16 + 2v(+1), n = ct*16 + mrow.
// Total: 3*4*8 fragments = 24576 dwords = 96 KB (L2-resident).
// ---------------------------------------------------------------------------
#define WP_DWORDS (3 * 4 * 8 * 32 * 8)

__global__ __launch_bounds__(256)
void pack_w_kernel(const float* __restrict__ Wq, const float* __restrict__ Wk,
                   const float* __restrict__ Wv, unsigned* __restrict__ Wp)
{
    const int i = blockIdx.x * blockDim.x + threadIdx.x;   // dword index
    if (i >= WP_DWORDS) return;
    const int v    = i & 7;
    const int lane = (i >> 3) & 31;
    const int ct   = (i >> 8) & 7;
    const int k32  = (i >> 11) & 3;
    const int m    = (i >> 13);
    const float* __restrict__ W = (m == 0) ? Wq : (m == 1) ? Wk : Wv;
    const int half = lane >> 4, mrow = lane & 15;
    const int kg   = k32 * 32 + half * 16 + 2 * v;
    const int coln = ct * 16 + mrow;
    Wp[i] = pk_bf16(W[(size_t)kg * HIDC + coln],
                    W[(size_t)(kg + 1) * HIDC + coln]);
}

// ---------------------------------------------------------------------------
// Kernel 1: fused QKV projection with WMMA (bf16 inputs, f32 accumulate)
// grid.x = ceil(N/16) node tiles, block = 128 threads (4 waves).
// Per tile & kk-step: 2x ds_load_b128 (A) + 2x global_load_b128 (B, pre-
// swizzled) + 1x v_wmma_f32_16x16x32_bf16 (4 unrolled over K=128).
// ---------------------------------------------------------------------------
__global__ __launch_bounds__(128)
void qkv_wmma_kernel(const float* __restrict__ h,
                     const unsigned* __restrict__ Wp,
                     const float* __restrict__ bq,
                     const float* __restrict__ bk,
                     const float* __restrict__ bv,
                     float* __restrict__ Q, float* __restrict__ K,
                     float* __restrict__ V, int n_nodes)
{
    // h tile as packed bf16 pairs: row-pitch 68 dwords (=272B, 16B aligned,
    // and (4*row + idx) % 64 is conflict-free across the 16 rows)
    __shared__ unsigned hP[16][68];

    const int tid  = threadIdx.x;
    const int wave = tid >> 5;
    const int lane = tid & 31;
    const int row0 = blockIdx.x * 16;

    // --- branch-free staging: thread -> (row = tid/8, 16 cols at (tid%8)*16)
    {
        const int sr = tid >> 3;
        const int sc = (tid & 7) * 16;
        int gr = row0 + sr;
        if (gr >= n_nodes) gr = n_nodes - 1;          // clamp; guarded at store
        const float4* __restrict__ hp =
            (const float4*)(h + (size_t)gr * HIDC + sc);
        const float4 f0 = hp[0], f1 = hp[1], f2 = hp[2], f3 = hp[3];
        uint4* wp = (uint4*)&hP[sr][sc >> 1];
        wp[0] = make_uint4(pk_bf16(f0.x, f0.y), pk_bf16(f0.z, f0.w),
                           pk_bf16(f1.x, f1.y), pk_bf16(f1.z, f1.w));
        wp[1] = make_uint4(pk_bf16(f2.x, f2.y), pk_bf16(f2.z, f2.w),
                           pk_bf16(f3.x, f3.y), pk_bf16(f3.z, f3.w));
    }
    __syncthreads();

    const int half = lane >> 4;   // K-half selector (contiguous k = 16*half + s in A and B)
    const int mrow = lane & 15;   // M (A frag) / N (B frag) index

    const bool full_tile = (row0 + 16) <= n_nodes;

    for (int t = wave; t < 24; t += 4) {
        const int m  = t >> 3;    // 0=Q, 1=K, 2=V
        const int ct = t & 7;     // 16-wide column tile
        const float* __restrict__ bb = (m == 0) ? bq : (m == 1) ? bk : bv;
        float*       __restrict__ O  = (m == 0) ? Q  : (m == 1) ? K  : V;

        const int coln = ct * 16 + mrow;
        const uint4* __restrict__ rp = (const uint4*)hP[mrow];
        // this wave's lane slice of the pre-swizzled B fragments for (m, ct);
        // consecutive kk-steps are 8 fragments (=2048 dwords = 512 uint4) apart
        const uint4* __restrict__ wfp =
            (const uint4*)(Wp + ((size_t)(m * 4 * 8 + ct) * 256) + lane * 8);

        v8f acc = {};
        #pragma unroll
        for (int kk32 = 0; kk32 < 4; ++kk32) {
            // A fragment 16x32: two 16B LDS reads, contiguous k = 16*half + s
            v8u apk;
            {
                const uint4 p0 = rp[kk32 * 4 + 2 * half];
                const uint4 p1 = rp[kk32 * 4 + 2 * half + 1];
                apk[0] = p0.x; apk[1] = p0.y; apk[2] = p0.z; apk[3] = p0.w;
                apk[4] = p1.x; apk[5] = p1.y; apk[6] = p1.z; apk[7] = p1.w;
            }
            const v16bf a = __builtin_bit_cast(v16bf, apk);

            // B fragment 32x16: two coalesced 16B global reads (pre-swizzled)
            v8u bpk;
            {
                const uint4 b0 = wfp[kk32 * 512];
                const uint4 b1 = wfp[kk32 * 512 + 1];
                bpk[0] = b0.x; bpk[1] = b0.y; bpk[2] = b0.z; bpk[3] = b0.w;
                bpk[4] = b1.x; bpk[5] = b1.y; bpk[6] = b1.z; bpk[7] = b1.w;
            }
            const v16bf b = __builtin_bit_cast(v16bf, bpk);

            acc = __builtin_amdgcn_wmma_f32_16x16x32_bf16(
                false, a, false, b, (short)0, acc, false, false);
        }

        // D layout: vgpr r holds row M = r + half*8, col N = mrow
        const float bias = bb[coln];
        if (full_tile) {
            #pragma unroll
            for (int r = 0; r < 8; ++r)
                O[(size_t)(row0 + r + half * 8) * HIDC + coln] = acc[r] + bias;
        } else {
            #pragma unroll
            for (int r = 0; r < 8; ++r) {
                const int grow = row0 + r + half * 8;
                if (grow < n_nodes) O[(size_t)grow * HIDC + coln] = acc[r] + bias;
            }
        }
    }
}

// ---------------------------------------------------------------------------
// Init: out = 0, m = -inf, denom = 0
// ---------------------------------------------------------------------------
__global__ void init_kernel(float* __restrict__ mbuf, float* __restrict__ denom,
                            float* __restrict__ out, int n_mh, int n_out)
{
    const int i = blockIdx.x * blockDim.x + threadIdx.x;
    if (i < n_out) out[i] = 0.0f;
    if (i < n_mh) { mbuf[i] = -__builtin_inff(); denom[i] = 0.0f; }
}

// float atomic max via the signed/unsigned integer-ordering trick
__device__ __forceinline__ void atomicMaxF(float* addr, float v)
{
    if (v >= 0.0f)
        atomicMax(reinterpret_cast<int*>(addr), __float_as_int(v));
    else
        atomicMin(reinterpret_cast<unsigned int*>(addr),
                  (unsigned int)__float_as_int(v));
}

// ---------------------------------------------------------------------------
// Pass A: per-(edge, head) score = dot(K[src], Q[dst]); segment max into m
// ---------------------------------------------------------------------------
__global__ __launch_bounds__(256)
void edge_score_kernel(const float* __restrict__ Q, const float* __restrict__ K,
                       const int* __restrict__ src, const int* __restrict__ dst,
                       float* __restrict__ score, float* __restrict__ mbuf, int E)
{
    const int i = blockIdx.x * blockDim.x + threadIdx.x;
    if (i >= E * NHEAD) return;
    const int e  = i >> 3;
    const int hh = i & 7;
    const int s = src[e], d = dst[e];

    const float4* __restrict__ kp =
        (const float4*)(K + (size_t)s * HIDC + hh * HD);
    const float4* __restrict__ qp =
        (const float4*)(Q + (size_t)d * HIDC + hh * HD);

    float acc = 0.0f;
    #pragma unroll
    for (int j = 0; j < 4; ++j) {
        const float4 kv = kp[j], qv = qp[j];
        acc += kv.x * qv.x + kv.y * qv.y + kv.z * qv.z + kv.w * qv.w;
    }
    score[i] = acc;
    atomicMaxF(&mbuf[(size_t)d * NHEAD + hh], acc);
}

// ---------------------------------------------------------------------------
// Pass B: p = exp(score - m[dst]); segment sum into denom
// ---------------------------------------------------------------------------
__global__ __launch_bounds__(256)
void edge_exp_kernel(const int* __restrict__ dst,
                     const float* __restrict__ mbuf,
                     float* __restrict__ p,          // in: score, out: exp()
                     float* __restrict__ denom, int E)
{
    const int i = blockIdx.x * blockDim.x + threadIdx.x;
    if (i >= E * NHEAD) return;
    const int e  = i >> 3;
    const int hh = i & 7;
    const int d = dst[e];
    const float pv = __expf(p[i] - mbuf[(size_t)d * NHEAD + hh]);
    p[i] = pv;
    atomicAdd(&denom[(size_t)d * NHEAD + hh], pv);
}

// ---------------------------------------------------------------------------
// Pass C: out[dst] += V[src] * (p / denom[dst])  (scatter-add, f32 atomics)
// ---------------------------------------------------------------------------
__global__ __launch_bounds__(256)
void edge_aggregate_kernel(const float* __restrict__ V,
                           const int* __restrict__ src, const int* __restrict__ dst,
                           const float* __restrict__ p, const float* __restrict__ denom,
                           float* __restrict__ out, int E)
{
    const int i = blockIdx.x * blockDim.x + threadIdx.x;
    if (i >= E * NHEAD) return;
    const int e  = i >> 3;
    const int hh = i & 7;
    const int s = src[e], d = dst[e];
    const float pn = p[i] / denom[(size_t)d * NHEAD + hh];

    const float4* __restrict__ vp =
        (const float4*)(V + (size_t)s * HIDC + hh * HD);
    float* __restrict__ op = out + (size_t)d * HIDC + hh * HD;
    #pragma unroll
    for (int j = 0; j < 4; ++j) {
        const float4 v = vp[j];
        atomicAdd(&op[j * 4 + 0], v.x * pn);
        atomicAdd(&op[j * 4 + 1], v.y * pn);
        atomicAdd(&op[j * 4 + 2], v.z * pn);
        atomicAdd(&op[j * 4 + 3], v.w * pn);
    }
}

// ---------------------------------------------------------------------------
extern "C" void kernel_launch(void* const* d_in, const int* in_sizes, int n_in,
                              void* d_out, int out_size, void* d_ws, size_t ws_size,
                              hipStream_t stream)
{
    const float* h   = (const float*)d_in[0];
    const int*   src = (const int*)  d_in[1];
    const int*   dst = (const int*)  d_in[2];
    const float* Wq  = (const float*)d_in[3];
    const float* bq  = (const float*)d_in[4];
    const float* Wk  = (const float*)d_in[5];
    const float* bk  = (const float*)d_in[6];
    const float* Wv  = (const float*)d_in[7];
    const float* bv  = (const float*)d_in[8];
    float* out = (float*)d_out;

    const int N = in_sizes[0] / HIDC;
    const int E = in_sizes[1];

    // Workspace layout: Q | K | V | p(score) | m | denom | Wp (pre-swizzled)
    float* ws = (float*)d_ws;
    const size_t nf = (size_t)N * HIDC;
    float*    Qb    = ws;
    float*    Kb    = Qb + nf;
    float*    Vb    = Kb + nf;
    float*    pbuf  = Vb + nf;                 // E * NHEAD
    float*    mbuf  = pbuf + (size_t)E * NHEAD;
    float*    denom = mbuf + (size_t)N * NHEAD;
    unsigned* Wpack = (unsigned*)(denom + (size_t)N * NHEAD);

    const int n_mh  = N * NHEAD;
    const int n_out = N * HIDC;
    const int n_eh  = E * NHEAD;

    // 0) pre-swizzle weights into WMMA B-fragment layout (bf16)
    {
        const int threads = 256;
        const int blocks = (WP_DWORDS + threads - 1) / threads;
        pack_w_kernel<<<blocks, threads, 0, stream>>>(Wq, Wk, Wv, Wpack);
    }
    // 1) zero/-inf init of reduction buffers and output
    {
        const int threads = 256;
        const int blocks = (n_out + threads - 1) / threads;
        init_kernel<<<blocks, threads, 0, stream>>>(mbuf, denom, out, n_mh, n_out);
    }
    // 2) QKV projection via WMMA
    {
        const int blocks = (N + 15) / 16;
        qkv_wmma_kernel<<<blocks, 128, 0, stream>>>(
            h, Wpack, bq, bk, bv, Qb, Kb, Vb, N);
    }
    // 3) edge scores + segment max
    {
        const int threads = 256;
        const int blocks = (n_eh + threads - 1) / threads;
        edge_score_kernel<<<blocks, threads, 0, stream>>>(Qb, Kb, src, dst, pbuf, mbuf, E);
    }
    // 4) exp + segment sum
    {
        const int threads = 256;
        const int blocks = (n_eh + threads - 1) / threads;
        edge_exp_kernel<<<blocks, threads, 0, stream>>>(dst, mbuf, pbuf, denom, E);
    }
    // 5) normalize + scatter-add messages
    {
        const int threads = 256;
        const int blocks = (n_eh + threads - 1) / threads;
        edge_aggregate_kernel<<<blocks, threads, 0, stream>>>(Vb, src, dst, pbuf, denom, out, E);
    }
}
